// MHPFGT_46849503265074
// MI455X (gfx1250) — compile-verified, matrix operand order checked by reading, not run on hardware
//
#include <hip/hip_runtime.h>
#include <hip/hip_bf16.h>

typedef __attribute__((ext_vector_type(16))) _Float16 v16h;
typedef __attribute__((ext_vector_type(8)))  float    v8f;

#define NFEAT   128
#define HIDDEN  128
#define NHEADS  4
#define HEADC   32
#define NCLS    10
#define KHOPS   3
#define CSTF    1e-5f
#define STATE_F 1408   /* 4*32*10 (M) + 4*32 (Kf) */
#define M_F     1280
#define MAXD    1024

// ---------------------------------------------------------------------------
// WMMA helpers: 16x16x32 f16 fragments per CDNA5 ISA 7.12.2
// A (16x32, 16-bit): lane(0..15)=row M, lane>>4 selects K-half (0..7 / 8..15,
// 16..23 / 24..31); VGPR v holds K pair: k = (v>>2)*16 + (lane>>4)*8 + (v&3)*2.
// B (32x16) uses the same pattern with lanes = column N over a K-major panel,
// so we stage B transposed (sB[n*128+k]) and reuse the same loader.
// D (16x16 f32): VGPR r -> M = r + 8*(lane>>4), N = lane&15.
// ---------------------------------------------------------------------------
__device__ __forceinline__ v16h frag16(const _Float16* __restrict__ panel, int lane, int kk) {
  const _Float16* r = panel + (lane & 15) * 128 + kk + ((lane >> 4) << 3);
  v16h f;
#pragma unroll
  for (int v = 0; v < 8; ++v) {
    int k = ((v >> 2) << 4) + ((v & 3) << 1);
    f[2 * v]     = r[k];
    f[2 * v + 1] = r[k + 1];
  }
  return f;
}

template <int NT>
__device__ __forceinline__ void gemm_panel(const _Float16* __restrict__ sArow,
                                           const _Float16* __restrict__ sB,
                                           int lane, v8f* acc) {
#pragma unroll
  for (int t = 0; t < NT; ++t)
#pragma unroll
    for (int i = 0; i < 8; ++i) acc[t][i] = 0.0f;
#pragma unroll
  for (int kk = 0; kk < 128; kk += 32) {
    v16h af = frag16(sArow, lane, kk);
#pragma unroll
    for (int t = 0; t < NT; ++t) {
      v16h bf = frag16(sB + t * 16 * 128, lane, kk);
      acc[t] = __builtin_amdgcn_wmma_f32_16x16x32_f16(
          /*neg_a=*/false, af, /*neg_b=*/false, bf,
          /*c_mod=*/(short)0, acc[t], /*reuse_a=*/false, /*reuse_b=*/false);
    }
  }
}

__device__ __forceinline__ void stage_bt(_Float16* __restrict__ sB,
                                         const float* __restrict__ W,
                                         int ncols, int npad, int tid) {
  // W is [128, ncols] (k-major). Store transposed+padded: sB[n*128 + k].
  for (int idx = tid; idx < 128 * npad; idx += 256) {
    int n = idx >> 7, k = idx & 127;
    sB[idx] = (n < ncols) ? (_Float16)W[k * ncols + n] : (_Float16)0.0f;
  }
}

// ---------------------------------------------------------------------------
// Kernel 1: fused  x=relu(feat@Wt+bt); Q=1+elu(x@Wq+bq); Kf=1+elu(x@Wk+bk);
//           V=x@Wv+bv.  128 rows per block, 8 waves, WMMA f16 (f32 acc).
// ---------------------------------------------------------------------------
__global__ __launch_bounds__(256) void k_qkv(
    const float* __restrict__ feat,
    const float* __restrict__ Wt, const float* __restrict__ bt,
    const float* __restrict__ Wq, const float* __restrict__ bq,
    const float* __restrict__ Wk, const float* __restrict__ bk,
    const float* __restrict__ Wv, const float* __restrict__ bv,
    float* __restrict__ Qg, float* __restrict__ Vg,
    float* __restrict__ stateA, int N) {
  __shared__ _Float16 sA[128 * 128];
  __shared__ _Float16 sB[128 * 128];
  const int tid = threadIdx.x, lane = tid & 31, wave = tid >> 5;
  const int node0 = blockIdx.x * 128;

  for (int idx = tid; idx < 128 * 128; idx += 256) {
    int r = idx >> 7, c = idx & 127;
    int node = node0 + r;
    sA[idx] = (_Float16)((node < N) ? feat[(size_t)node * NFEAT + c] : 0.0f);
  }
  stage_bt(sB, Wt, 128, 128, tid);
  __syncthreads();

  const _Float16* sArow = sA + (wave * 16) * 128;
  const int lrow  = wave * 16 + ((lane >> 4) << 3);
  const int nbase = lane & 15;
  v8f acc[8];

  // ---- GEMM1: x = relu(feat @ Wt + bt), write f16 back to own sA rows ----
  gemm_panel<8>(sArow, sB, lane, acc);
#pragma unroll
  for (int t = 0; t < 8; ++t) {
    int n = t * 16 + nbase;
    float b = bt[n];
#pragma unroll
    for (int r = 0; r < 8; ++r) {
      float z = acc[t][r] + b;
      sA[(lrow + r) * 128 + n] = (_Float16)(z > 0.0f ? z : 0.0f);
    }
  }
  __syncthreads();

  // ---- GEMM2: Q = 1 + elu(x @ Wq + bq) ----
  stage_bt(sB, Wq, 128, 128, tid);
  __syncthreads();
  gemm_panel<8>(sArow, sB, lane, acc);
#pragma unroll
  for (int t = 0; t < 8; ++t) {
    int n = t * 16 + nbase;
    float b = bq[n];
#pragma unroll
    for (int r = 0; r < 8; ++r) {
      int node = node0 + lrow + r;
      if (node < N) {
        float z = acc[t][r] + b;
        Qg[(size_t)node * HIDDEN + n] = (z > 0.0f) ? (1.0f + z) : __expf(z);
      }
    }
  }
  __syncthreads();

  // ---- GEMM3: Kf = 1 + elu(x @ Wk + bk)  -> state[., 1280..1407] ----
  stage_bt(sB, Wk, 128, 128, tid);
  __syncthreads();
  gemm_panel<8>(sArow, sB, lane, acc);
#pragma unroll
  for (int t = 0; t < 8; ++t) {
    int n = t * 16 + nbase;
    float b = bk[n];
#pragma unroll
    for (int r = 0; r < 8; ++r) {
      int node = node0 + lrow + r;
      if (node < N) {
        float z = acc[t][r] + b;
        stateA[(size_t)node * STATE_F + M_F + n] = (z > 0.0f) ? (1.0f + z) : __expf(z);
      }
    }
  }
  __syncthreads();

  // ---- GEMM4: V = x @ Wv + bv  (40 cols, padded to 48) ----
  stage_bt(sB, Wv, 40, 48, tid);
  __syncthreads();
  v8f accv[3];
  gemm_panel<3>(sArow, sB, lane, accv);
#pragma unroll
  for (int t = 0; t < 3; ++t) {
    int n = t * 16 + nbase;
    if (n < 40) {
      float b = bv[n];
#pragma unroll
      for (int r = 0; r < 8; ++r) {
        int node = node0 + lrow + r;
        if (node < N) Vg[(size_t)node * 40 + n] = accv[t][r] + b;
      }
    }
  }
}

// ---------------------------------------------------------------------------
// Kernel 2: state M0 = Kf ⊗ V,  out = hopwise[0] * sum_h V
// ---------------------------------------------------------------------------
__global__ __launch_bounds__(128) void k_init(
    float* __restrict__ stateA, const float* __restrict__ Vg,
    const float* __restrict__ hopwise, float* __restrict__ out, int N) {
  const int n = blockIdx.x;
  const int tid = threadIdx.x;
  __shared__ float sK[128];
  __shared__ float sV[40];
  const size_t sb = (size_t)n * STATE_F;
  sK[tid] = stateA[sb + M_F + tid];
  if (tid < 40) sV[tid] = Vg[(size_t)n * 40 + tid];
  __syncthreads();
  for (int idx = tid; idx < M_F; idx += 128) {
    int h = idx / 320, rem = idx - h * 320;
    int i = rem / 10, j = rem - i * 10;
    stateA[sb + idx] = sK[h * 32 + i] * sV[h * 10 + j];
  }
  if (tid < 10) {
    float hw0 = hopwise[0];
    out[(size_t)n * 10 + tid] =
        hw0 * (sV[tid] + sV[10 + tid] + sV[20 + tid] + sV[30 + tid]);
  }
}

// ---------------------------------------------------------------------------
// CSR build (by destination = col)
// ---------------------------------------------------------------------------
__global__ void k_deg(const long long* __restrict__ ei, int* __restrict__ deg, int E) {
  int e = blockIdx.x * 256 + threadIdx.x;
  if (e < E) atomicAdd(&deg[(int)ei[(size_t)E + e]], 1);
}

__global__ __launch_bounds__(1024) void k_scan(const int* __restrict__ deg,
                                               int* __restrict__ offs, int N) {
  __shared__ int s[1024];
  __shared__ int sbase;
  const int tid = threadIdx.x;
  if (tid == 0) sbase = 0;
  __syncthreads();
  for (int start = 0; start < N; start += 1024) {
    int i = start + tid;
    int v = (i < N) ? deg[i] : 0;
    s[tid] = v;
    __syncthreads();
    for (int off = 1; off < 1024; off <<= 1) {
      int t = (tid >= off) ? s[tid - off] : 0;
      __syncthreads();
      s[tid] += t;
      __syncthreads();
    }
    int incl = s[tid];
    int total = s[1023];
    int base = sbase;
    __syncthreads();
    if (i < N) offs[i] = base + incl - v;
    if (tid == 0) sbase = base + total;
    __syncthreads();
  }
  if (tid == 0) offs[N] = sbase;
}

__global__ void k_cursor(const int* __restrict__ offs, int* __restrict__ cursor, int N) {
  int i = blockIdx.x * 256 + threadIdx.x;
  if (i < N) cursor[i] = offs[i];
}

__global__ void k_fill(const long long* __restrict__ ei, int* __restrict__ cursor,
                       int* __restrict__ csr, int E) {
  int e = blockIdx.x * 256 + threadIdx.x;
  if (e < E) {
    int c = (int)ei[(size_t)E + e];
    int p = atomicAdd(&cursor[c], 1);
    csr[p] = (int)ei[e];
  }
}

// ---------------------------------------------------------------------------
// Hop kernel: one workgroup per destination node.
//   new_state[n] = sum over incoming src of old_state[src]   (gather, no atomics)
//   H = Q·M, C = Q·Kf + eps, out[n] += sum_h gamma_h * H/C   (fused epilogue)
// ---------------------------------------------------------------------------
__global__ __launch_bounds__(256) void k_hop(
    const float* __restrict__ st_in, float* __restrict__ st_out,
    const int* __restrict__ offs, const int* __restrict__ csr,
    const float* __restrict__ Qg, const float* __restrict__ hopwise,
    const float* __restrict__ headwise, float* __restrict__ out,
    int hop, int N) {
  const int n = blockIdx.x;
  const int tid = threadIdx.x;
  const int s0 = offs[n], s1 = offs[n + 1];
  const int deg = s1 - s0;

  __shared__ int   ls[MAXD];
  __shared__ int   ls2[MAXD];
  __shared__ float s[STATE_F];
  __shared__ float hc[40];

  float a0 = 0.f, a1 = 0.f, a2 = 0.f, a3 = 0.f, a4 = 0.f, a5 = 0.f;

  if (deg <= MAXD) {
    // deterministic order: rank-sort the (small) neighbor list
    for (int i = tid; i < deg; i += 256) ls[i] = csr[s0 + i];
    __syncthreads();
    for (int i = tid; i < deg; i += 256) {
      int v = ls[i], r = 0;
      for (int k = 0; k < deg; ++k) {
        int u = ls[k];
        r += (u < v) || (u == v && k < i);
      }
      ls2[r] = v;
    }
    __syncthreads();
    for (int e = 0; e < deg; ++e) {
      const float* sp = st_in + (size_t)ls2[e] * STATE_F;
      a0 += sp[tid];
      a1 += sp[tid + 256];
      a2 += sp[tid + 512];
      a3 += sp[tid + 768];
      a4 += sp[tid + 1024];
      if (tid < 128) a5 += sp[tid + 1280];
    }
  } else {
    for (int e = 0; e < deg; ++e) {
      const float* sp = st_in + (size_t)csr[s0 + e] * STATE_F;
      a0 += sp[tid];
      a1 += sp[tid + 256];
      a2 += sp[tid + 512];
      a3 += sp[tid + 768];
      a4 += sp[tid + 1024];
      if (tid < 128) a5 += sp[tid + 1280];
    }
  }

  if (st_out) {  // last hop skips the state write
    float* dp = st_out + (size_t)n * STATE_F;
    dp[tid] = a0;
    dp[tid + 256] = a1;
    dp[tid + 512] = a2;
    dp[tid + 768] = a3;
    dp[tid + 1024] = a4;
    if (tid < 128) dp[tid + 1280] = a5;
  }

  s[tid] = a0;
  s[tid + 256] = a1;
  s[tid + 512] = a2;
  s[tid + 768] = a3;
  s[tid + 1024] = a4;
  if (tid < 128) s[tid + 1280] = a5;
  __syncthreads();

  if (tid < 40) {
    const int h = tid / 10, j = tid - h * 10;
    const float* q  = Qg + (size_t)n * HIDDEN + h * HEADC;
    const float* m  = s + h * (HEADC * NCLS);
    const float* kf = s + M_F + h * HEADC;
    float Hv = 0.f, C = CSTF;
#pragma unroll
    for (int i = 0; i < HEADC; ++i) {
      Hv += q[i] * m[i * NCLS + j];
      C  += q[i] * kf[i];
    }
    // gamma = hopwise[hop+1] * softmax_over_heads(headwise)[h, hop]
    float den = 0.f;
#pragma unroll
    for (int hh = 0; hh < NHEADS; ++hh) den += __expf(headwise[hh * KHOPS + hop]);
    float g = hopwise[hop + 1] * __expf(headwise[h * KHOPS + hop]) / den;
    hc[tid] = g * Hv / C;
  }
  __syncthreads();
  if (tid < 10)
    out[(size_t)n * 10 + tid] += hc[tid] + hc[10 + tid] + hc[20 + tid] + hc[30 + tid];
}

// ---------------------------------------------------------------------------
extern "C" void kernel_launch(void* const* d_in, const int* in_sizes, int n_in,
                              void* d_out, int out_size, void* d_ws, size_t ws_size,
                              hipStream_t stream) {
  const float*     feat     = (const float*)d_in[0];
  const long long* ei       = (const long long*)d_in[1];
  const float*     Wt       = (const float*)d_in[2];
  const float*     bt       = (const float*)d_in[3];
  const float*     Wq       = (const float*)d_in[4];
  const float*     bq       = (const float*)d_in[5];
  const float*     Wk       = (const float*)d_in[6];
  const float*     bk       = (const float*)d_in[7];
  const float*     Wv       = (const float*)d_in[8];
  const float*     bv       = (const float*)d_in[9];
  const float*     hopwise  = (const float*)d_in[10];
  const float*     headwise = (const float*)d_in[11];
  float*           out      = (float*)d_out;

  const int N = in_sizes[0] / NFEAT;
  const int E = in_sizes[1] / 2;

  char* w = (char*)d_ws;
  auto carve = [&](size_t bytes) -> void* {
    void* p = (void*)w;
    w += (bytes + 255) & ~(size_t)255;
    return p;
  };
  float* stateA = (float*)carve((size_t)N * STATE_F * sizeof(float));
  float* stateB = (float*)carve((size_t)N * STATE_F * sizeof(float));
  float* Qg     = (float*)carve((size_t)N * HIDDEN * sizeof(float));
  float* Vg     = (float*)carve((size_t)N * 40 * sizeof(float));
  int*   deg    = (int*)carve((size_t)N * sizeof(int));
  int*   offs   = (int*)carve((size_t)(N + 1) * sizeof(int));
  int*   cursor = (int*)carve((size_t)N * sizeof(int));
  int*   csr    = (int*)carve((size_t)E * sizeof(int));

  // CSR-by-destination build
  hipMemsetAsync(deg, 0, (size_t)N * sizeof(int), stream);
  k_deg<<<(E + 255) / 256, 256, 0, stream>>>(ei, deg, E);
  k_scan<<<1, 1024, 0, stream>>>(deg, offs, N);
  k_cursor<<<(N + 255) / 256, 256, 0, stream>>>(offs, cursor, N);
  k_fill<<<(E + 255) / 256, 256, 0, stream>>>(ei, cursor, csr, E);

  // Fused WMMA GEMMs + epilogues
  k_qkv<<<(N + 127) / 128, 256, 0, stream>>>(feat, Wt, bt, Wq, bq, Wk, bk, Wv, bv,
                                             Qg, Vg, stateA, N);
  // M0 = Kf ⊗ V, out init
  k_init<<<N, 128, 0, stream>>>(stateA, Vg, hopwise, out, N);

  // 3 hops (ping-pong state; last hop skips the state write)
  k_hop<<<N, 256, 0, stream>>>(stateA, stateB, offs, csr, Qg, hopwise, headwise, out, 0, N);
  k_hop<<<N, 256, 0, stream>>>(stateB, stateA, offs, csr, Qg, hopwise, headwise, out, 1, N);
  k_hop<<<N, 256, 0, stream>>>(stateA, nullptr, offs, csr, Qg, hopwise, headwise, out, 2, N);
}
